// NFTnetworkBlock_68006512165217
// MI455X (gfx1250) — matrix operand order repaired and checked by reading, not compiled, not run on hardware
//
#include <hip/hip_runtime.h>

typedef __bf16 bf16;
typedef __attribute__((ext_vector_type(16))) __bf16 v16bf;
typedef __attribute__((ext_vector_type(8)))  float  v8f;

#define BB   4
#define TT   8192
#define CC   256
#define EE   8
#define DIN  1024
#define NTOK (BB*TT)          // 32768
#define H1P  264              // LDS pitch for h1 (bank rotation)

union Frag { uint4 q[2]; v16bf v; };

__device__ __forceinline__ v8f wmma_bf16(v16bf a, v16bf b, v8f c) {
  return __builtin_amdgcn_wmma_f32_16x16x32_bf16(false, a, false, b, (short)0, c, false, false);
}
__device__ __forceinline__ float red32(float v) {
  v += __shfl_xor(v, 1, 32);  v += __shfl_xor(v, 2, 32);
  v += __shfl_xor(v, 4, 32);  v += __shfl_xor(v, 8, 32);
  v += __shfl_xor(v, 16, 32);
  return v;
}
__device__ __forceinline__ float red16(float v) {   // reduce across lanes sharing (lane>>4)
  v += __shfl_xor(v, 1, 32);  v += __shfl_xor(v, 2, 32);
  v += __shfl_xor(v, 4, 32);  v += __shfl_xor(v, 8, 32);
  return v;
}
// Branchless erf-based gelu (Abramowitz-Stegun 7.1.26, |err| < 1.5e-7):
// avoids the branchy libm erff that forces EXEC save/restore next to WMMA.
__device__ __forceinline__ float gelu_f(float x) {
  float a = fabsf(x) * 0.70710678118654752f;       // |x|/sqrt(2)
  float t = __builtin_amdgcn_rcpf(__builtin_fmaf(0.3275911f, a, 1.0f));
  float p = t * (0.254829592f +
            t * (-0.284496736f +
            t * (1.421413741f +
            t * (-1.453152027f +
            t * 1.061405429f))));
  float er = __builtin_fmaf(-p, __expf(-a * a), 1.0f);
  er = copysignf(er, x);
  return 0.5f * x * (1.0f + er);
}

// ---------- conversions ----------
__global__ void k_cvt_x(const float* __restrict__ x, bf16* __restrict__ xb, int n) {
  for (int i = blockIdx.x * blockDim.x + threadIdx.x; i < n; i += gridDim.x * blockDim.x)
    xb[i] = (bf16)x[i];
}
// W1xT[e][c][d] = W1[e][d][c], d<256 (the x-part only), bf16
__global__ void k_cvt_w1(const float* __restrict__ W1, bf16* __restrict__ W1xT) {
  int i = blockIdx.x * blockDim.x + threadIdx.x;
  if (i >= EE * CC * CC) return;
  int e = i >> 16, r = i & 65535, c = r >> 8, d = r & 255;
  W1xT[i] = (bf16)W1[(size_t)(e * DIN + d) * CC + c];
}
// W2T[e][f][c] = W2[e][c][f], bf16
__global__ void k_cvt_w2(const float* __restrict__ W2, bf16* __restrict__ W2T) {
  int i = blockIdx.x * blockDim.x + threadIdx.x;
  if (i >= EE * CC * CC) return;
  int e = i >> 16, r = i & 65535, f = r >> 8, c = r & 255;
  W2T[i] = (bf16)W2[(size_t)(e * CC + c) * CC + f];
}

// ---------- masked LN statistics (pass 1: accumulate) ----------
__global__ void k_stats(const float* __restrict__ x, const int* __restrict__ mask,
                        const float* __restrict__ lnw, const float* __restrict__ lnb,
                        float* __restrict__ gsum, float* __restrict__ gsq,
                        float* __restrict__ gcnt) {
  const int b = blockIdx.x >> 5;          // 32 blocks / batch
  const int chunk = blockIdx.x & 31;      // 256 tokens / block
  const int wave = threadIdx.x >> 5, lane = threadIdx.x & 31;
  float lw[8], lb[8];
#pragma unroll
  for (int j = 0; j < 8; ++j) { int c = lane + 32 * j; lw[j] = lnw[c]; lb[j] = lnb[c]; }
  float hs[8] = {0,0,0,0,0,0,0,0}, hq[8] = {0,0,0,0,0,0,0,0}, cnt = 0.f;
  for (int it = 0; it < 32; ++it) {
    int t = chunk * 256 + wave * 32 + it;
    const float* xp = x + ((size_t)(b * TT + t)) * CC;
    float v[8], s = 0.f;
#pragma unroll
    for (int j = 0; j < 8; ++j) { v[j] = xp[lane + 32 * j]; s += v[j]; }
    s = red32(s);
    float mu = s * (1.0f / 256.0f), q = 0.f;
#pragma unroll
    for (int j = 0; j < 8; ++j) { float d = v[j] - mu; q += d * d; }
    q = red32(q);
    float inv = rsqrtf(q * (1.0f / 256.0f) + 1e-5f);
    if (mask[b * TT + t]) {
#pragma unroll
      for (int j = 0; j < 8; ++j) {
        float h = (v[j] - mu) * inv * lw[j] + lb[j];
        hs[j] += h; hq[j] += h * h;
      }
      if (lane == 0) cnt += 1.f;
    }
  }
  __shared__ float red[8][256];
  __shared__ float rc[8];
  if (lane == 0) rc[wave] = cnt;
#pragma unroll
  for (int j = 0; j < 8; ++j) red[wave][lane + 32 * j] = hs[j];
  __syncthreads();
  { float t2 = 0.f;
#pragma unroll
    for (int w = 0; w < 8; ++w) t2 += red[w][threadIdx.x];
    atomicAdd(&gsum[b * CC + threadIdx.x], t2); }
  __syncthreads();
#pragma unroll
  for (int j = 0; j < 8; ++j) red[wave][lane + 32 * j] = hq[j];
  __syncthreads();
  { float t2 = 0.f;
#pragma unroll
    for (int w = 0; w < 8; ++w) t2 += red[w][threadIdx.x];
    atomicAdd(&gsq[b * CC + threadIdx.x], t2); }
  if (threadIdx.x == 0) {
    float t2 = 0.f;
#pragma unroll
    for (int w = 0; w < 8; ++w) t2 += rc[w];
    atomicAdd(&gcnt[b], t2);
  }
}

// ---------- finalize stats: stats[b][0:256]=mean, [256:512]=std_u, [512:768]=var_u ----------
__global__ void k_finalize(const float* __restrict__ gsum, const float* __restrict__ gsq,
                           const float* __restrict__ gcnt, float* __restrict__ stats) {
  int i = blockIdx.x * blockDim.x + threadIdx.x;
  if (i >= BB * CC) return;
  int b = i >> 8, c = i & 255;
  float n = fmaxf(gcnt[b], 1.0f);
  float mean = gsum[i] / n;
  float varb = fmaxf(gsq[i] / n - mean * mean, 0.0f);
  float corr = n / fmaxf(n - 1.0f, 1e-9f);
  float varu = (n > 1.0f) ? varb * corr : varb;
  float stdu = fmaxf(sqrtf(varu), 1e-9f);
  stats[b * 768 + c] = mean;
  stats[b * 768 + 256 + c] = stdu;
  stats[b * 768 + 512 + c] = varu;
}

// ---------- fold stats through W1 rows 256..1023 into per-(b,e) bias ----------
__global__ void k_bias1(const float* __restrict__ W1, const float* __restrict__ b1,
                        const float* __restrict__ stats, float* __restrict__ bias1) {
  int i = blockIdx.x * blockDim.x + threadIdx.x;
  if (i >= BB * EE * CC) return;
  int b = i >> 11, r = i & 2047, e = r >> 8, c = r & 255;
  float s = b1[e * CC + c];
  const float* st = stats + b * 768;
  for (int j = 0; j < 768; ++j)
    s += st[j] * W1[(size_t)(e * DIN + 256 + j) * CC + c];
  bias1[i] = s;
}

// ---------- u[e][c] = W2[e][c][:]·We[e][:] ; be2[e] = b2[e]·We[e] + be[e] ----------
__global__ void k_uvec(const float* __restrict__ W2, const float* __restrict__ We,
                       const float* __restrict__ b2, const float* __restrict__ be,
                       float* __restrict__ u, float* __restrict__ be2) {
  int i = blockIdx.x * blockDim.x + threadIdx.x;
  if (i < EE * CC) {
    int e = i >> 8, c = i & 255;
    float s = 0.f;
    const float* w = W2 + (size_t)(e * CC + c) * CC;
    const float* wv = We + e * CC;
    for (int f = 0; f < CC; ++f) s += w[f] * wv[f];
    u[i] = s;
  }
  if (i < EE) {
    float s = 0.f;
    const float* bv = b2 + i * CC;
    const float* wv = We + i * CC;
    for (int f = 0; f < CC; ++f) s += bv[f] * wv[f];
    be2[i] = s + be[i];
  }
}

// ---------- fused expert MLP ensemble + energy softmax mixture ----------
__global__ void __launch_bounds__(256)
k_main(const bf16* __restrict__ xb, const int* __restrict__ mask,
       const float* __restrict__ bias1, const bf16* __restrict__ W1xT,
       const bf16* __restrict__ W2T, const float* __restrict__ u,
       const float* __restrict__ be2, const float* __restrict__ b2,
       const float* __restrict__ log_beta, const float* __restrict__ prior,
       float* __restrict__ out) {
  extern __shared__ char smem[];
  bf16* h1s   = (bf16*)smem;                                    // [8][16][H1P]
  float* energ = (float*)(smem + (size_t)EE * 16 * H1P * 2);    // [8][16]
  float* probs = energ + 128;                                   // [16][8]
  float* mfv   = probs + 128;                                   // [16]

  const int tid = threadIdx.x;
  const int t0g = blockIdx.x * 16;
  const int b = t0g >> 13;                 // /8192
  if (tid < 128) energ[tid] = 0.f;
  if (tid < 16)  mfv[tid] = mask[t0g + tid] ? 1.0f : 0.0f;
  __syncthreads();

  const int wave = tid >> 5, lane = tid & 31;
  const int m16 = lane & 15, half = lane >> 4, kb = half * 8;

  // Preload A fragments of x tile (shared across all experts / column tiles)
  Frag afr[8];
#pragma unroll
  for (int kk = 0; kk < 8; ++kk) {
    const uint4* pa = (const uint4*)(xb + (size_t)(t0g + m16) * CC + kk * 32 + kb);
    afr[kk].q[0] = pa[0]; afr[kk].q[1] = pa[2];     // K chunks [kb..kb+7], [kb+16..kb+23]
  }

  // ---- Pass A: GEMM1 (K=256) + bias1 + gelu -> h1 (LDS, bf16); energies via u ----
  for (int e = 0; e < EE; ++e) {
#pragma unroll
    for (int rep = 0; rep < 2; ++rep) {
      const int c = (wave + 8 * rep) * 16 + m16;
      const float bias = bias1[(size_t)((b * EE + e) * CC + c)];
      v8f acc;
#pragma unroll
      for (int i = 0; i < 8; ++i) acc[i] = bias;
      const bf16* wrow = W1xT + (size_t)(e * CC + c) * CC;
#pragma unroll
      for (int kk = 0; kk < 8; ++kk) {
        Frag bfr;
        const uint4* pb = (const uint4*)(wrow + kk * 32 + half * 8);
        bfr.q[0] = pb[0]; bfr.q[1] = pb[2];
        acc = wmma_bf16(afr[kk].v, bfr.v, acc);
      }
      const float uc = u[e * CC + c];
#pragma unroll
      for (int i = 0; i < 8; ++i) {
        float g = gelu_f(acc[i]);
        h1s[(e * 16 + half * 8 + i) * H1P + c] = (bf16)g;
        float part = red16(g * uc);
        if (m16 == 0) atomicAdd(&energ[e * 16 + half * 8 + i], part);
      }
    }
  }
  __syncthreads();

  // ---- softmax over experts per token ----
  if (tid < 16) {
    if (mfv[tid] > 0.f) {
      const float beta = __expf(log_beta[0]);
      float nb[EE], mx = -3.402823466e38f;
#pragma unroll
      for (int e = 0; e < EE; ++e) {
        nb[e] = -beta * (energ[e * 16 + tid] + be2[e] + prior[e]);
        mx = fmaxf(mx, nb[e]);
      }
      float s = 0.f;
#pragma unroll
      for (int e = 0; e < EE; ++e) { nb[e] = __expf(nb[e] - mx); s += nb[e]; }
      float inv = 1.0f / s;
#pragma unroll
      for (int e = 0; e < EE; ++e) probs[tid * 8 + e] = nb[e] * inv;
    } else {
#pragma unroll
      for (int e = 0; e < EE; ++e) probs[tid * 8 + e] = 0.f;
    }
  }
  __syncthreads();

  // ---- scale h1 by probs in LDS (so Pass B is a single chained accumulation) ----
  for (int idx = tid; idx < EE * 16 * CC; idx += 256) {
    int e = idx >> 12, t = (idx >> 8) & 15, c2 = idx & 255;
    int a = (e * 16 + t) * H1P + c2;
    h1s[a] = (bf16)((float)h1s[a] * probs[t * 8 + e]);
  }
  __syncthreads();

  // ---- Pass B: out = sum_e (p_e*h1_e) @ W2[e] + sum_e p_e*b2[e], masked ----
#pragma unroll
  for (int rep = 0; rep < 2; ++rep) {
    const int c = (wave + 8 * rep) * 16 + m16;
    float b2r[EE];
#pragma unroll
    for (int e = 0; e < EE; ++e) b2r[e] = b2[e * CC + c];
    v8f acc;
#pragma unroll
    for (int i = 0; i < 8; ++i) {
      int t = half * 8 + i;
      float s = 0.f;
#pragma unroll
      for (int e = 0; e < EE; ++e) s += probs[t * 8 + e] * b2r[e];
      acc[i] = s;
    }
    for (int e = 0; e < EE; ++e) {
      const bf16* wrow = W2T + (size_t)(e * CC + c) * CC;
      const bf16* hrow = h1s + (e * 16 + m16) * H1P;
#pragma unroll
      for (int kk = 0; kk < 8; ++kk) {
        Frag af, bfr;
        const uint4* pa = (const uint4*)(hrow + kk * 32 + kb);
        af.q[0] = pa[0]; af.q[1] = pa[2];
        const uint4* pb = (const uint4*)(wrow + kk * 32 + half * 8);
        bfr.q[0] = pb[0]; bfr.q[1] = pb[2];
        acc = wmma_bf16(af.v, bfr.v, acc);
      }
    }
#pragma unroll
    for (int i = 0; i < 8; ++i) {
      int t = half * 8 + i;
      out[(size_t)(t0g + t) * CC + c] = acc[i] * mfv[t];
    }
  }
}

extern "C" void kernel_launch(void* const* d_in, const int* in_sizes, int n_in,
                              void* d_out, int out_size, void* d_ws, size_t ws_size,
                              hipStream_t stream) {
  (void)in_sizes; (void)n_in; (void)out_size; (void)ws_size;
  const float* x    = (const float*)d_in[0];
  const int*   mask = (const int*)d_in[1];
  const float* lnw  = (const float*)d_in[2];
  const float* lnb  = (const float*)d_in[3];
  const float* W1   = (const float*)d_in[4];
  const float* b1   = (const float*)d_in[5];
  const float* W2   = (const float*)d_in[6];
  const float* b2   = (const float*)d_in[7];
  const float* We   = (const float*)d_in[8];
  const float* be   = (const float*)d_in[9];
  const float* logb = (const float*)d_in[10];
  const float* prior= (const float*)d_in[11];
  float* out = (float*)d_out;

  char* ws = (char*)d_ws;
  size_t off = 0;
  bf16*  xb    = (bf16*)(ws + off);  off += (size_t)NTOK * CC * 2;        // 16 MB
  bf16*  W1xT  = (bf16*)(ws + off);  off += (size_t)EE * CC * CC * 2;     // 1 MB
  bf16*  W2T   = (bf16*)(ws + off);  off += (size_t)EE * CC * CC * 2;     // 1 MB
  float* gsum  = (float*)(ws + off); off += BB * CC * 4;                  // 4 KB
  float* gsq   = (float*)(ws + off); off += BB * CC * 4;                  // 4 KB
  float* gcnt  = (float*)(ws + off); off += 256;
  float* stats = (float*)(ws + off); off += BB * 768 * 4;                 // 12 KB
  float* bias1 = (float*)(ws + off); off += BB * EE * CC * 4;             // 32 KB
  float* uvec  = (float*)(ws + off); off += EE * CC * 4;                  // 8 KB
  float* be2   = (float*)(ws + off); off += 256;

  // zero the stat accumulators (contiguous gsum|gsq|gcnt region)
  hipMemsetAsync(gsum, 0, BB * CC * 4 * 2 + 256, stream);

  k_cvt_x <<<4096, 256, 0, stream>>>(x, xb, NTOK * CC);
  k_cvt_w1<<<(EE * CC * CC + 255) / 256, 256, 0, stream>>>(W1, W1xT);
  k_cvt_w2<<<(EE * CC * CC + 255) / 256, 256, 0, stream>>>(W2, W2T);
  k_stats <<<BB * 32, 256, 0, stream>>>(x, mask, lnw, lnb, gsum, gsq, gcnt);
  k_finalize<<<(BB * CC + 255) / 256, 256, 0, stream>>>(gsum, gsq, gcnt, stats);
  k_bias1 <<<(BB * EE * CC + 255) / 256, 256, 0, stream>>>(W1, b1, stats, bias1);
  k_uvec  <<<(EE * CC + 255) / 256, 256, 0, stream>>>(W2, We, b2, be, uvec, be2);

  const int SMEM = EE * 16 * H1P * 2 + 128 * 4 + 128 * 4 + 16 * 4;  // 68,672 B
  hipFuncSetAttribute((const void*)k_main, hipFuncAttributeMaxDynamicSharedMemorySize, SMEM);
  k_main<<<NTOK / 16, 256, SMEM, stream>>>(xb, mask, bias1, W1xT, W2T, uvec, be2,
                                           b2, logb, prior, out);
}
